// SameVocabRewardModel_46926812676819
// MI455X (gfx1250) — matrix operand order repaired and checked by reading, not compiled, not run on hardware
//
#include <hip/hip_runtime.h>
#include <hip/hip_bf16.h>
#include <math.h>

// ---------------- model dims ----------------
#define NL   2
#define DM   2048
#define NH   16
#define DH   128
#define FFD  8192
#define NB   2
#define SQ   1024
#define NT   (NB * SQ)          // 2048 tokens

// GEMM tiling
#define BM   256
#define BN   128
#define BK   32

typedef __bf16 bf16_t;
typedef __attribute__((ext_vector_type(16))) __bf16 v16bf;
typedef __attribute__((ext_vector_type(8)))  __bf16 bf16x8;
typedef __attribute__((ext_vector_type(8)))  float  v8f;
typedef __attribute__((ext_vector_type(4)))  float  f32x4;

// ---------------- fp32 -> bf16 bulk convert (weights) ----------------
__global__ void k_f2bf(const float* __restrict__ src, bf16_t* __restrict__ dst, long n) {
    long i = ((long)blockIdx.x * 256 + threadIdx.x) * 4;
    if (i + 3 < n) {
        f32x4 v = *(const f32x4*)(src + i);
        dst[i + 0] = (bf16_t)v.x;
        dst[i + 1] = (bf16_t)v.y;
        dst[i + 2] = (bf16_t)v.z;
        dst[i + 3] = (bf16_t)v.w;
    }
}

// ---------------- embedding gather ----------------
__global__ void k_embed(const int* __restrict__ ids, const float* __restrict__ emb,
                        float* __restrict__ x) {
    int i = blockIdx.x * 256 + threadIdx.x;          // over NT*DM
    int t = i / DM, d = i % DM;
    x[i] = emb[(long)ids[t] * DM + d];
}

// ---------------- layernorm (block per token) ----------------
template <typename OUT>
__global__ void k_layernorm(const float* __restrict__ x, const float* __restrict__ g,
                            const float* __restrict__ b, OUT* __restrict__ out) {
    int t = blockIdx.x;
    const float* row = x + (size_t)t * DM;
    __shared__ float red[256];
    int tid = threadIdx.x;

    float s = 0.f;
    for (int d = tid; d < DM; d += 256) s += row[d];
    red[tid] = s; __syncthreads();
    for (int o = 128; o > 0; o >>= 1) { if (tid < o) red[tid] += red[tid + o]; __syncthreads(); }
    float mean = red[0] * (1.0f / DM);
    __syncthreads();

    float v = 0.f;
    for (int d = tid; d < DM; d += 256) { float dv = row[d] - mean; v += dv * dv; }
    red[tid] = v; __syncthreads();
    for (int o = 128; o > 0; o >>= 1) { if (tid < o) red[tid] += red[tid + o]; __syncthreads(); }
    float rstd = rsqrtf(red[0] * (1.0f / DM) + 1e-5f);

    for (int d = tid; d < DM; d += 256)
        out[(size_t)t * DM + d] = (OUT)((row[d] - mean) * rstd * g[d] + b[d]);
}

// ---------------- generic batched bf16 WMMA GEMM ----------------
// C[z] = epilogue(A[z] @ B[z]); A:[M,K] bf16, B:[K,N] bf16, row-major.
// Block = 256 thr (8 waves), tile BMxBN=256x128, wave tile 64x64, K-step 32,
// double-buffered LDS with register prefetch (1 barrier / K-step).
enum { EPI_NONE = 0, EPI_ATTN = 1, EPI_GELU = 2, EPI_RES = 3 };

__device__ __forceinline__ float gelu_tanh(float v) {
    return 0.5f * v * (1.0f + tanhf(0.7978845608028654f * (v + 0.044715f * v * v * v)));
}

template <typename TC, int EPI, int CAUSAL>
__global__ __launch_bounds__(256)
void k_gemm_wmma(const bf16_t* __restrict__ Ag, const bf16_t* __restrict__ Bg,
                 TC* __restrict__ Cg, const float* __restrict__ Resg,
                 int M, int N, int K, int lda, int ldb, int ldc,
                 long sA, long sB, long csB, long csH, int HC,
                 float scale, const int* __restrict__ amask, int Hheads) {
    __shared__ __align__(16) bf16_t As[2][BM][BK];
    __shared__ __align__(16) bf16_t Bs[2][BK][BN];

    const int z = blockIdx.z;
    const bf16_t* A = Ag + (long)z * sA;
    const bf16_t* B = Bg + (long)z * sB;
    const long coff = (long)(z / HC) * csB + (long)(z % HC) * csH;
    TC* C = Cg + coff;
    const float* R = (EPI == EPI_RES) ? (Resg + coff) : nullptr;

    const int bm = blockIdx.y * BM, bn = blockIdx.x * BN;
    const int tid = threadIdx.x, lane = tid & 31, w = tid >> 5;
    const int wm = (w & 3) * 64, wn = (w >> 2) * 64;

    if constexpr (EPI == EPI_ATTN) {
        // tile entirely above the causal diagonal: all masked
        if (bn > bm + (BM - 1)) {
            for (int i = tid; i < BM * BN; i += 256) {
                int r = i / BN, c = i % BN;
                C[(long)(bm + r) * ldc + (bn + c)] = (TC)(-1e9f);
            }
            return;
        }
    }
    const int Kend = CAUSAL ? ((K < bm + BM) ? K : (bm + BM)) : K;

    // tile-load thread mapping
    const int ar = tid;                              // A: row, 32 cols (4 x bf16x8)
    const int br = tid >> 3, bc0 = (tid & 7) * 16;   // B: row, 16 cols (2 x bf16x8)

    v8f acc[4][4];
#pragma unroll
    for (int i = 0; i < 4; ++i)
#pragma unroll
        for (int j = 0; j < 4; ++j) acc[i][j] = (v8f)0.0f;

    const int half = lane >> 4, mrow = lane & 15;

    // preload tile k0=0 into LDS buffer 0
    {
        const bf16_t* asrc = A + (long)(bm + ar) * lda;
        const bf16_t* bsrc = B + (long)br * ldb + bn + bc0;
#pragma unroll
        for (int j = 0; j < 4; ++j) *(bf16x8*)&As[0][ar][j * 8] = *(const bf16x8*)(asrc + j * 8);
        *(bf16x8*)&Bs[0][br][bc0]     = *(const bf16x8*)(bsrc);
        *(bf16x8*)&Bs[0][br][bc0 + 8] = *(const bf16x8*)(bsrc + 8);
    }
    __syncthreads();

    for (int k0 = 0; k0 < Kend; k0 += BK) {
        const int buf = (k0 / BK) & 1;
        const bool more = (k0 + BK) < Kend;

        // register-prefetch next tile
        bf16x8 apf[4], bpf[2];
        if (more) {
            const bf16_t* asrc = A + (long)(bm + ar) * lda + (k0 + BK);
            const bf16_t* bsrc = B + (long)(k0 + BK + br) * ldb + bn + bc0;
#pragma unroll
            for (int j = 0; j < 4; ++j) apf[j] = *(const bf16x8*)(asrc + j * 8);
            bpf[0] = *(const bf16x8*)(bsrc);
            bpf[1] = *(const bf16x8*)(bsrc + 8);
        }

        // A fragments: lane (l%16)=M row; elems 0..7 = K[half*8..+7], 8..15 = K[16+half*8..+7]
        v16bf af[4];
#pragma unroll
        for (int mt = 0; mt < 4; ++mt) {
            const bf16_t* rp = &As[buf][wm + mt * 16 + mrow][0];
            bf16x8 lo = *(const bf16x8*)(rp + half * 8);
            bf16x8 hi = *(const bf16x8*)(rp + 16 + half * 8);
            af[mt] = __builtin_shufflevector(lo, hi, 0, 1, 2, 3, 4, 5, 6, 7,
                                                     8, 9, 10, 11, 12, 13, 14, 15);
        }
        // B fragments: lane (l%16)=N col; elems 0..7 = K[half*8..+7], 8..15 = K[16+half*8..+7]
        v16bf bfr[4];
#pragma unroll
        for (int nt = 0; nt < 4; ++nt) {
            const bf16_t* cp = &Bs[buf][0][wn + nt * 16 + mrow];
#pragma unroll
            for (int s = 0; s < 8; ++s) {
                bfr[nt][s]     = cp[(half * 8 + s) * BN];
                bfr[nt][s + 8] = cp[(16 + half * 8 + s) * BN];
            }
        }
#pragma unroll
        for (int mt = 0; mt < 4; ++mt)
#pragma unroll
            for (int nt = 0; nt < 4; ++nt)
                acc[mt][nt] = __builtin_amdgcn_wmma_f32_16x16x32_bf16(
                    false, af[mt], false, bfr[nt], (short)0, acc[mt][nt], false, false);

        // stage prefetched tile into the other LDS buffer
        if (more) {
            const int nb = buf ^ 1;
#pragma unroll
            for (int j = 0; j < 4; ++j) *(bf16x8*)&As[nb][ar][j * 8] = apf[j];
            *(bf16x8*)&Bs[nb][br][bc0]     = bpf[0];
            *(bf16x8*)&Bs[nb][br][bc0 + 8] = bpf[1];
        }
        __syncthreads();
    }

    // store: C/D layout: lane(0-15)->N=lane, VGPR v -> M=v (+8 for lanes 16-31)
    const int lcol = lane & 15, lhalf = lane >> 4;
#pragma unroll
    for (int mt = 0; mt < 4; ++mt)
#pragma unroll
        for (int nt = 0; nt < 4; ++nt)
#pragma unroll
            for (int v = 0; v < 8; ++v) {
                int row = bm + wm + mt * 16 + v + 8 * lhalf;
                int col = bn + wn + nt * 16 + lcol;
                float val = acc[mt][nt][v];
                if constexpr (EPI == EPI_ATTN) {
                    int b = z / Hheads;
                    bool ok = (col <= row) && (amask[(long)b * N + col] != 0);
                    val = ok ? val * scale : -1e9f;
                } else if constexpr (EPI == EPI_GELU) {
                    val = gelu_tanh(val);
                } else if constexpr (EPI == EPI_RES) {
                    val += R[(long)row * ldc + col];
                }
                C[(long)row * ldc + col] = (TC)val;
            }
}

// ---------------- split qkv -> Q, K^T, V (per-head bf16) ----------------
__global__ void k_qkv_split(const bf16_t* __restrict__ qkv, bf16_t* __restrict__ Q,
                            bf16_t* __restrict__ Kt, bf16_t* __restrict__ V) {
    int i = blockIdx.x * 256 + threadIdx.x;          // over NT*DM
    int t = i / DM, c = i % DM;
    int h = c / DH, dh = c % DH;
    int b = t / SQ, s = t % SQ;
    long zb = (long)(b * NH + h);
    const bf16_t* src = qkv + (long)t * (3 * DM);
    Q [(zb * SQ + s) * DH + dh] = src[c];
    Kt[(zb * DH + dh) * SQ + s] = src[DM + c];
    V [(zb * SQ + s) * DH + dh] = src[2 * DM + c];
}

// ---------------- row softmax (block per row), fp32 -> bf16 ----------------
__global__ void k_softmax_rows(const float* __restrict__ sc, bf16_t* __restrict__ pr) {
    long row = blockIdx.x;                           // NB*NH*SQ rows
    const float* r = sc + row * (long)SQ;
    bf16_t* o = pr + row * (long)SQ;
    __shared__ float red[256];
    int tid = threadIdx.x;

    float mx = -3.0e38f;
    for (int j = tid; j < SQ; j += 256) mx = fmaxf(mx, r[j]);
    red[tid] = mx; __syncthreads();
    for (int off = 128; off > 0; off >>= 1) { if (tid < off) red[tid] = fmaxf(red[tid], red[tid + off]); __syncthreads(); }
    mx = red[0];
    __syncthreads();

    float s = 0.f;
    for (int j = tid; j < SQ; j += 256) s += __expf(r[j] - mx);
    red[tid] = s; __syncthreads();
    for (int off = 128; off > 0; off >>= 1) { if (tid < off) red[tid] += red[tid + off]; __syncthreads(); }
    float inv = 1.0f / red[0];

    for (int j = tid; j < SQ; j += 256) o[j] = (bf16_t)(__expf(r[j] - mx) * inv);
}

// ---------------- masked mean pool ----------------
__global__ void k_pool(const float* __restrict__ hs, const int* __restrict__ amask,
                       float* __restrict__ pooled) {
    int d = blockIdx.x * 256 + threadIdx.x;
    int b = blockIdx.y;
    float acc = 0.f, cnt = 0.f;
    for (int s = 0; s < SQ; ++s) {
        float m = (float)amask[b * SQ + s];
        acc += hs[((long)(b * SQ + s)) * DM + d] * m;
        cnt += m;
    }
    pooled[b * DM + d] = acc / fmaxf(cnt, 1e-9f);
}

// ---------------- classifier head (block per batch) ----------------
__global__ void k_classifier(const float* __restrict__ pooled, const float* __restrict__ Wc1,
                             const float* __restrict__ bc1, const float* __restrict__ Wc2,
                             const float* __restrict__ bc2, float* __restrict__ out) {
    int b = blockIdx.x, tid = threadIdx.x;
    __shared__ float tval[DM];
    __shared__ float red0[256], red1[256];
    const float* p = pooled + b * DM;
    for (int j = tid; j < DM; j += 256) {
        float acc = bc1[j];
        for (int i = 0; i < DM; ++i) acc += p[i] * Wc1[(long)i * DM + j];
        tval[j] = tanhf(acc);
    }
    __syncthreads();
    float l0 = 0.f, l1 = 0.f;
    for (int j = tid; j < DM; j += 256) { l0 += tval[j] * Wc2[j * 2]; l1 += tval[j] * Wc2[j * 2 + 1]; }
    red0[tid] = l0; red1[tid] = l1; __syncthreads();
    for (int off = 128; off > 0; off >>= 1) {
        if (tid < off) { red0[tid] += red0[tid + off]; red1[tid] += red1[tid + off]; }
        __syncthreads();
    }
    if (tid == 0) {
        float a = red0[0] + bc2[0], c = red1[0] + bc2[1];
        float m = fmaxf(a, c);
        float e0 = __expf(a - m), e1 = __expf(c - m);
        out[b] = e1 / (e0 + e1);
    }
}

// ---------------- host-side GEMM wrapper ----------------
template <typename TC, int EPI, int CAUSAL = 0>
static void gemm(hipStream_t st, const bf16_t* A, const bf16_t* B, TC* C, const float* Res,
                 int M, int N, int K, int lda, int ldb, int ldc, int batches,
                 long sA, long sB, long csB, long csH, int HC,
                 float scale, const int* amask, int Hheads) {
    dim3 g(N / BN, M / BM, batches);
    k_gemm_wmma<TC, EPI, CAUSAL><<<g, 256, 0, st>>>(A, B, C, Res, M, N, K, lda, ldb, ldc,
                                                    sA, sB, csB, csH, HC, scale, amask, Hheads);
}

static void f2bf(hipStream_t st, const float* src, bf16_t* dst, long n) {
    long blocks = (n + 1023) / 1024;                 // 4 elems / thread
    k_f2bf<<<(int)blocks, 256, 0, st>>>(src, dst, n);
}

extern "C" void kernel_launch(void* const* d_in, const int* in_sizes, int n_in,
                              void* d_out, int out_size, void* d_ws, size_t ws_size,
                              hipStream_t stream) {
    const int*   ids    = (const int*)d_in[0];
    const int*   amask  = (const int*)d_in[1];
    const float* emb    = (const float*)d_in[2];
    const float* ln1_g  = (const float*)d_in[3];
    const float* ln1_b  = (const float*)d_in[4];
    const float* Wqkv   = (const float*)d_in[5];
    const float* Wo     = (const float*)d_in[6];
    const float* ln2_g  = (const float*)d_in[7];
    const float* ln2_b  = (const float*)d_in[8];
    const float* W1     = (const float*)d_in[9];
    const float* W2     = (const float*)d_in[10];
    const float* lnf_g  = (const float*)d_in[11];
    const float* lnf_b  = (const float*)d_in[12];
    const float* Wc1    = (const float*)d_in[13];
    const float* bc1    = (const float*)d_in[14];
    const float* Wc2    = (const float*)d_in[15];
    const float* bc2    = (const float*)d_in[16];
    float* out = (float*)d_out;

    char* wp = (char*)d_ws;
    size_t off = 0;
    auto alloc = [&](size_t bytes) -> void* {
        void* p = wp + off;
        off = (off + bytes + 255) & ~(size_t)255;
        return p;
    };
    // activations
    float*  x      = (float*) alloc((size_t)NT * DM * 4);
    bf16_t* hb     = (bf16_t*)alloc((size_t)NT * DM * 2);
    bf16_t* qkv    = (bf16_t*)alloc((size_t)NT * 3 * DM * 2);
    bf16_t* Q      = (bf16_t*)alloc((size_t)NB * NH * SQ * DH * 2);
    bf16_t* Kt     = (bf16_t*)alloc((size_t)NB * NH * SQ * DH * 2);
    bf16_t* V      = (bf16_t*)alloc((size_t)NB * NH * SQ * DH * 2);
    float*  scores = (float*) alloc((size_t)NB * NH * SQ * SQ * 4);
    bf16_t* probs  = (bf16_t*)alloc((size_t)NB * NH * SQ * SQ * 2);
    bf16_t* ob     = (bf16_t*)alloc((size_t)NT * DM * 2);
    bf16_t* a1     = (bf16_t*)alloc((size_t)NT * FFD * 2);
    float*  hs     = (float*) alloc((size_t)NT * DM * 4);
    float*  pooled = (float*) alloc((size_t)NB * DM * 4);
    // bf16 weight copies
    bf16_t* wqkv_b = (bf16_t*)alloc((size_t)NL * DM * 3 * DM * 2);
    bf16_t* wo_b   = (bf16_t*)alloc((size_t)NL * DM * DM * 2);
    bf16_t* w1_b   = (bf16_t*)alloc((size_t)NL * DM * FFD * 2);
    bf16_t* w2_b   = (bf16_t*)alloc((size_t)NL * FFD * DM * 2);
    (void)ws_size; (void)in_sizes; (void)n_in; (void)out_size;

    const float scale = 1.0f / sqrtf((float)DH);

    // convert weights to bf16 once per launch
    f2bf(stream, Wqkv, wqkv_b, (long)NL * DM * 3 * DM);
    f2bf(stream, Wo,   wo_b,   (long)NL * DM * DM);
    f2bf(stream, W1,   w1_b,   (long)NL * DM * FFD);
    f2bf(stream, W2,   w2_b,   (long)NL * FFD * DM);

    // x = emb[ids]
    k_embed<<<(NT * DM) / 256, 256, 0, stream>>>(ids, emb, x);

    for (int l = 0; l < NL; ++l) {
        const bf16_t* wqkv = wqkv_b + (long)l * DM * 3 * DM;
        const bf16_t* wo   = wo_b   + (long)l * DM * DM;
        const bf16_t* w1   = w1_b   + (long)l * DM * FFD;
        const bf16_t* w2   = w2_b   + (long)l * FFD * DM;

        // h = LN1(x)  (bf16)
        k_layernorm<bf16_t><<<NT, 256, 0, stream>>>(x, ln1_g + l * DM, ln1_b + l * DM, hb);
        // qkv = h @ Wqkv  (bf16 out)
        gemm<bf16_t, EPI_NONE>(stream, hb, wqkv, qkv, nullptr,
            NT, 3 * DM, DM, DM, 3 * DM, 3 * DM, 1, 0, 0, 0, 0, 1, 0.f, nullptr, 1);
        // split into per-head Q, K^T, V
        k_qkv_split<<<(NT * DM) / 256, 256, 0, stream>>>(qkv, Q, Kt, V);
        // scores = scale * Q @ K^T + causal/pad bias (batched over B*H, masked tiles skipped)
        gemm<float, EPI_ATTN>(stream, Q, Kt, scores, nullptr,
            SQ, SQ, DH, DH, SQ, SQ, NB * NH,
            (long)SQ * DH, (long)DH * SQ, (long)SQ * SQ, 0, 1, scale, amask, NH);
        // probs = softmax(scores) (bf16)
        k_softmax_rows<<<NB * NH * SQ, 256, 0, stream>>>(scores, probs);
        // o[b, q, h*DH + dh] = probs @ V  (batched, strided C, causal K-clip)
        gemm<bf16_t, EPI_NONE, 1>(stream, probs, V, ob, nullptr,
            SQ, DH, SQ, SQ, DH, DM, NB * NH,
            (long)SQ * SQ, (long)SQ * DH, (long)SQ * DM, (long)DH, NH, 0.f, nullptr, 1);
        // x = x + o @ Wo
        gemm<float, EPI_RES>(stream, ob, wo, x, x,
            NT, DM, DM, DM, DM, DM, 1, 0, 0, 0, 0, 1, 0.f, nullptr, 1);
        // h = LN2(x) (bf16)
        k_layernorm<bf16_t><<<NT, 256, 0, stream>>>(x, ln2_g + l * DM, ln2_b + l * DM, hb);
        // a1 = gelu(h @ W1)  (bf16)
        gemm<bf16_t, EPI_GELU>(stream, hb, w1, a1, nullptr,
            NT, FFD, DM, DM, FFD, FFD, 1, 0, 0, 0, 0, 1, 0.f, nullptr, 1);
        // x = x + a1 @ W2
        gemm<float, EPI_RES>(stream, a1, w2, x, x,
            NT, DM, FFD, FFD, DM, DM, 1, 0, 0, 0, 0, 1, 0.f, nullptr, 1);
    }

    // hs = LN_f(x) (fp32), masked mean pool, classifier head
    k_layernorm<float><<<NT, 256, 0, stream>>>(x, lnf_g, lnf_b, hs);
    {
        dim3 g(DM / 256, NB);
        k_pool<<<g, 256, 0, stream>>>(hs, amask, pooled);
    }
    k_classifier<<<NB, 256, 0, stream>>>(pooled, Wc1, bc1, Wc2, bc2, out);
}
